// HybridCrossAttentionClassifier_61959198212625
// MI455X (gfx1250) — matrix operand order repaired and checked by reading, not compiled, not run on hardware
//
#include <hip/hip_runtime.h>
#include <hip/hip_bf16.h>

typedef __attribute__((ext_vector_type(16))) _Float16 v16h;
typedef __attribute__((ext_vector_type(8)))  float    v8f;

#define B_SZ 8
#define L_SZ 1024
#define C_IN 12
#define D_MODEL 128
#define D_STATE 16
#define D_CONV 4
#define D_INNER 256
#define DT_RANK 8
#define NHEAD 4
#define HDIM 32
#define LSTM_H 128
#define FC_H 128
#define NCLS 5
#define DBL_W 64   /* x_proj output padded from 40 -> 64 cols */

static __device__ __forceinline__ float sigf(float x) { return 1.f / (1.f + __expf(-x)); }
static __device__ __forceinline__ float siluf(float x) { return x / (1.f + __expf(-x)); }
static __device__ __forceinline__ float softplusf(float x) { return (x > 20.f) ? x : log1pf(__expf(x)); }

static __device__ __forceinline__ v8f wmma_f16(v16h a, v16h b, v8f c) {
    return __builtin_amdgcn_wmma_f32_16x16x32_f16(false, a, false, b, (short)0, c, false, false);
}

// ---------------------------------------------------------------------------
// Weight pack: W (K x N, f32, row-major) -> Wt (Npad x Kpad, f16, transposed,
// zero padded).  Run once per launch; makes the GEMM hot loop branch-free.
// ---------------------------------------------------------------------------
__global__ void pack_w_kernel(const float* __restrict__ W, _Float16* __restrict__ Wt,
                              int K, int N, int Kpad, int Npad)
{
    int idx = blockIdx.x * 256 + threadIdx.x;
    if (idx >= Npad * Kpad) return;
    int n = idx / Kpad, k = idx % Kpad;
    float v = (k < K && n < N) ? W[(size_t)k * N + n] : 0.f;
    Wt[idx] = (_Float16)v;
}

__global__ void f16cvt_kernel(const float* __restrict__ in, _Float16* __restrict__ out, int n)
{
    int i = blockIdx.x * 256 + threadIdx.x;
    if (i < n) out[i] = (_Float16)in[i];
}

// V (B,L,128) f32 -> Vt [(b*128+hd)][L] f16  (per-head transposed)
__global__ void vtrans_kernel(const float* __restrict__ V, _Float16* __restrict__ Vt)
{
    int idx = blockIdx.x * 256 + threadIdx.x; // total B*128*1024
    if (idx >= B_SZ * D_MODEL * L_SZ) return;
    int l = idx & (L_SZ - 1);
    int r = idx >> 10;       // b*128 + hd
    int b = r >> 7;
    int hd = r & 127;
    Vt[idx] = (_Float16)V[((size_t)b * L_SZ + l) * D_MODEL + hd];
}

// ---------------------------------------------------------------------------
// Branch-free WMMA GEMM: C[M,N] = act(A[M,K] @ Wt^T + bias + resid)
// A: f32 row-major (lda), Wt: f16 (N x K) packed.  M mult 16, N mult 64,
// K mult 32.  One wave computes a 16x64 strip: A-frag loaded once per k-step,
// 4 WMMAs against 4 B-fragments -> 4x A reuse, 2x WMMA density.
// ---------------------------------------------------------------------------
__global__ void __launch_bounds__(32)
wmma_gemm_kernel(const float* __restrict__ A, const _Float16* __restrict__ Wt,
                 const float* __restrict__ bias, const float* __restrict__ resid,
                 float* __restrict__ C, int K, int lda, int ldc, int act)
{
    int lane = threadIdx.x;
    int lh = lane >> 4, l15 = lane & 15;
    int col0 = blockIdx.x * 64, row0 = blockIdx.y * 16;

    const float*    ap = A  + (size_t)(row0 + l15) * lda + lh * 16;
    const _Float16* wp = Wt + (size_t)(col0 + l15) * K   + lh * 16;
    const size_t wstep = (size_t)16 * K;   // 16 columns forward in packed layout

    v8f acc0 = {}, acc1 = {}, acc2 = {}, acc3 = {};
    for (int k0 = 0; k0 < K; k0 += 32) {
        __builtin_prefetch(ap + k0 + 32, 0, 3);   // next A tile -> global_prefetch
        float4 a0 = *(const float4*)(ap + k0);
        float4 a1 = *(const float4*)(ap + k0 + 4);
        float4 a2 = *(const float4*)(ap + k0 + 8);
        float4 a3 = *(const float4*)(ap + k0 + 12);
        v16h af;
        af[0] = (_Float16)a0.x; af[1] = (_Float16)a0.y; af[2] = (_Float16)a0.z; af[3] = (_Float16)a0.w;
        af[4] = (_Float16)a1.x; af[5] = (_Float16)a1.y; af[6] = (_Float16)a1.z; af[7] = (_Float16)a1.w;
        af[8] = (_Float16)a2.x; af[9] = (_Float16)a2.y; af[10] = (_Float16)a2.z; af[11] = (_Float16)a2.w;
        af[12] = (_Float16)a3.x; af[13] = (_Float16)a3.y; af[14] = (_Float16)a3.z; af[15] = (_Float16)a3.w;
        v16h b0 = *(const v16h*)(wp + k0);
        v16h b1 = *(const v16h*)(wp + wstep + k0);
        v16h b2 = *(const v16h*)(wp + 2 * wstep + k0);
        v16h b3 = *(const v16h*)(wp + 3 * wstep + k0);
        acc0 = wmma_f16(af, b0, acc0);
        acc1 = wmma_f16(af, b1, acc1);
        acc2 = wmma_f16(af, b2, acc2);
        acc3 = wmma_f16(af, b3, acc3);
    }

    v8f accs[4] = {acc0, acc1, acc2, acc3};
#pragma unroll
    for (int j = 0; j < 4; ++j) {
        int cc = col0 + j * 16 + l15;
        float bv = bias ? bias[cc] : 0.f;
#pragma unroll
        for (int i = 0; i < 8; ++i) {
            int rr = row0 + 8 * lh + i;
            float v = accs[j][i] + bv;
            if (resid) v += resid[(size_t)rr * ldc + cc];
            if (act == 1) v = siluf(v);
            else if (act == 2) v = softplusf(v);
            C[(size_t)rr * ldc + cc] = v;
        }
    }
}

// ---------------------------------------------------------------------------
__global__ void mproj_kernel(const float* __restrict__ x, const float* __restrict__ w,
                             const float* __restrict__ b, float* __restrict__ u)
{
    int row = blockIdx.x, t = threadIdx.x;
    const float* xr = x + (size_t)row * C_IN;
    float acc = b[t];
#pragma unroll
    for (int k = 0; k < C_IN; ++k) acc += xr[k] * w[k * D_MODEL + t];
    u[(size_t)row * D_MODEL + t] = acc;
}

__global__ void conv_kernel(const float* __restrict__ xz, const float* __restrict__ cw,
                            const float* __restrict__ cb, float* __restrict__ xc)
{
    int idx = blockIdx.x * 256 + threadIdx.x;
    if (idx >= B_SZ * L_SZ * D_INNER) return;
    int d = idx & (D_INNER - 1);
    int bl = idx >> 8;
    int l = bl & (L_SZ - 1);
    int b = bl >> 10;
    float acc = cb[d];
#pragma unroll
    for (int k = 0; k < D_CONV; ++k) {
        int ls = l + k - (D_CONV - 1);
        if (ls >= 0) acc += xz[((size_t)(b * L_SZ + ls)) * (2 * D_INNER) + d] * cw[d * D_CONV + k];
    }
    xc[(size_t)idx] = siluf(acc);
}

// Selective scan: one thread per (b,d); fuses SiLU(z) gate.
__global__ void scan_kernel(const float* __restrict__ delta, const float* __restrict__ xc,
                            const float* __restrict__ dbl, const float* __restrict__ xz,
                            const float* __restrict__ A_log, const float* __restrict__ Dp,
                            float* __restrict__ yz)
{
    int t = blockIdx.x * blockDim.x + threadIdx.x;
    if (t >= B_SZ * D_INNER) return;
    int b = t / D_INNER, d = t % D_INNER;

    float A[D_STATE], h[D_STATE];
#pragma unroll
    for (int s = 0; s < D_STATE; ++s) { A[s] = -__expf(A_log[d * D_STATE + s]); h[s] = 0.f; }
    float dp = Dp[d];

    for (int l = 0; l < L_SZ; ++l) {
        size_t r = (size_t)b * L_SZ + l;
        float de = delta[r * D_INNER + d];
        float xcv = xc[r * D_INNER + d];
        const float* dr = dbl + r * DBL_W;
        float zv = xz[r * (2 * D_INNER) + D_INNER + d];
        float y = 0.f;
#pragma unroll
        for (int s = 0; s < D_STATE; ++s) {
            float da = __expf(de * A[s]);
            h[s] = da * h[s] + de * dr[DT_RANK + s] * xcv;
            y += h[s] * dr[DT_RANK + D_STATE + s];
        }
        y = (y + dp * xcv) * siluf(zv);
        yz[r * D_INNER + d] = y;
    }
}

__global__ void lstm_kernel(const float* __restrict__ x, const float* __restrict__ w_ih,
                            const float* __restrict__ w_hh, const float* __restrict__ bias,
                            float* __restrict__ H_L)
{
    __shared__ float h_s[LSTM_H];
    __shared__ float c_s[LSTM_H];
    __shared__ float g_s[4 * LSTM_H];
    int b = blockIdx.x, t = threadIdx.x;
    if (t < LSTM_H) { h_s[t] = 0.f; c_s[t] = 0.f; }
    __syncthreads();
    for (int l = 0; l < L_SZ; ++l) {
        const float* xt = x + ((size_t)b * L_SZ + l) * C_IN;
        float acc = bias[t];
#pragma unroll
        for (int k = 0; k < C_IN; ++k) acc += xt[k] * w_ih[k * 4 * LSTM_H + t];
#pragma unroll 4
        for (int k = 0; k < LSTM_H; ++k) acc += h_s[k] * w_hh[k * 4 * LSTM_H + t];
        g_s[t] = acc;
        __syncthreads();
        if (t < LSTM_H) {
            float ig = sigf(g_s[t]);
            float fg = sigf(g_s[LSTM_H + t]);
            float gg = tanhf(g_s[2 * LSTM_H + t]);
            float og = sigf(g_s[3 * LSTM_H + t]);
            float c = fg * c_s[t] + ig * gg;
            float h = og * tanhf(c);
            c_s[t] = c; h_s[t] = h;
            H_L[((size_t)b * L_SZ + l) * LSTM_H + t] = h;
        }
        __syncthreads();
    }
}

// ---------------------------------------------------------------------------
// Flash attention: one wave per (b, head, 16-query tile); f16-staged K and
// per-head transposed f16 V -> all fragments are contiguous vector loads.
// ---------------------------------------------------------------------------
__global__ void __launch_bounds__(32)
attn_kernel(const float* __restrict__ Q, const _Float16* __restrict__ Kh,
            const _Float16* __restrict__ Vt, float* __restrict__ O)
{
    __shared__ float sc[16][36];      // padded for 16B-aligned re-reads
    __shared__ float alpha_s[16];
    __shared__ float lsum_s[16];
    int lane = threadIdx.x;
    int lh = lane >> 4, l15 = lane & 15;
    int q0 = blockIdx.x * 16;
    int h = blockIdx.y, b = blockIdx.z;
    size_t base = (size_t)b * L_SZ * D_MODEL + h * HDIM;
    int bh = b * NHEAD + h;
    const float scale = 0.17677669529663687f; // 1/sqrt(32)

    v16h qa;
    {
        const float* qp = Q + base + (size_t)(q0 + l15) * D_MODEL + lh * 16;
        float4 a0 = *(const float4*)(qp);
        float4 a1 = *(const float4*)(qp + 4);
        float4 a2 = *(const float4*)(qp + 8);
        float4 a3 = *(const float4*)(qp + 12);
        qa[0] = (_Float16)(a0.x * scale); qa[1] = (_Float16)(a0.y * scale);
        qa[2] = (_Float16)(a0.z * scale); qa[3] = (_Float16)(a0.w * scale);
        qa[4] = (_Float16)(a1.x * scale); qa[5] = (_Float16)(a1.y * scale);
        qa[6] = (_Float16)(a1.z * scale); qa[7] = (_Float16)(a1.w * scale);
        qa[8] = (_Float16)(a2.x * scale); qa[9] = (_Float16)(a2.y * scale);
        qa[10] = (_Float16)(a2.z * scale); qa[11] = (_Float16)(a2.w * scale);
        qa[12] = (_Float16)(a3.x * scale); qa[13] = (_Float16)(a3.y * scale);
        qa[14] = (_Float16)(a3.z * scale); qa[15] = (_Float16)(a3.w * scale);
    }
    v8f o0 = {}, o1 = {};
    float m_run = -1e30f, l_run = 0.f;

    for (int c = 0; c < L_SZ; c += 32) {
        v16h kb0 = *(const v16h*)(Kh + base + (size_t)(c + l15) * D_MODEL + lh * 16);
        v16h kb1 = *(const v16h*)(Kh + base + (size_t)(c + 16 + l15) * D_MODEL + lh * 16);
        v8f s0 = {}, s1 = {};
        s0 = wmma_f16(qa, kb0, s0);
        s1 = wmma_f16(qa, kb1, s1);
#pragma unroll
        for (int i = 0; i < 8; ++i) {
            sc[i + 8 * lh][l15] = s0[i];
            sc[i + 8 * lh][16 + l15] = s1[i];
        }
        __syncthreads();
        {   // online softmax: 2 lanes per row, 16 cols each, combine via shfl
            int row = l15, c0 = lh * 16;
            float cm = -1e30f;
#pragma unroll
            for (int j = 0; j < 16; ++j) cm = fmaxf(cm, sc[row][c0 + j]);
            cm = fmaxf(cm, __shfl_xor(cm, 16, 32));
            float m_new = fmaxf(m_run, cm);
            float al = __expf(m_run - m_new);
            float rs = 0.f;
#pragma unroll
            for (int j = 0; j < 16; ++j) {
                float p = __expf(sc[row][c0 + j] - m_new);
                sc[row][c0 + j] = p;
                rs += p;
            }
            rs += __shfl_xor(rs, 16, 32);
            l_run = l_run * al + rs;
            m_run = m_new;
            if (lane < 16) alpha_s[row] = al;
        }
        __syncthreads();
#pragma unroll
        for (int i = 0; i < 8; ++i) {
            float al = alpha_s[i + 8 * lh];
            o0[i] *= al; o1[i] *= al;
        }
        v16h pa;
        {
            const float* pr = &sc[l15][lh * 16];
#pragma unroll
            for (int e = 0; e < 8; ++e) pa[e] = (_Float16)pr[e];
#pragma unroll
            for (int e = 0; e < 8; ++e) pa[8 + e] = (_Float16)pr[8 + e];
        }
        v16h vb0 = *(const v16h*)(Vt + ((size_t)(bh * HDIM + l15)) * L_SZ + c + lh * 16);
        v16h vb1 = *(const v16h*)(Vt + ((size_t)(bh * HDIM + 16 + l15)) * L_SZ + c + lh * 16);
        o0 = wmma_f16(pa, vb0, o0);
        o1 = wmma_f16(pa, vb1, o1);
        __syncthreads();
    }
    if (lane < 16) lsum_s[lane] = l_run;
    __syncthreads();
#pragma unroll
    for (int i = 0; i < 8; ++i) {
        int r = i + 8 * lh;
        float inv = 1.f / lsum_s[r];
        size_t orow = base + (size_t)(q0 + r) * D_MODEL;
        O[orow + l15] = o0[i] * inv;
        O[orow + 16 + l15] = o1[i] * inv;
    }
}

// ---------------------------------------------------------------------------
__global__ void ln_kernel(const float* __restrict__ in, const float* __restrict__ g,
                          const float* __restrict__ bt, float* __restrict__ out)
{
    __shared__ float red[D_MODEL];
    int row = blockIdx.x, t = threadIdx.x;
    float v = in[(size_t)row * D_MODEL + t];
    red[t] = v;
    __syncthreads();
    for (int s = D_MODEL / 2; s > 0; s >>= 1) { if (t < s) red[t] += red[t + s]; __syncthreads(); }
    float mean = red[0] * (1.f / D_MODEL);
    __syncthreads();
    float dv = v - mean;
    red[t] = dv * dv;
    __syncthreads();
    for (int s = D_MODEL / 2; s > 0; s >>= 1) { if (t < s) red[t] += red[t + s]; __syncthreads(); }
    float var = red[0] * (1.f / D_MODEL);
    out[(size_t)row * D_MODEL + t] = dv * rsqrtf(var + 1e-5f) * g[t] + bt[t];
}

__global__ void pool_kernel(const float* __restrict__ H_Mu, const float* __restrict__ H_Lu,
                            float* __restrict__ fused)
{
    int b = blockIdx.x, t = threadIdx.x; // 256 threads
    const float* src = (t < D_MODEL) ? H_Mu : H_Lu;
    int c = t & (D_MODEL - 1);
    float s = 0.f;
    for (int l = 0; l < L_SZ; ++l) s += src[((size_t)b * L_SZ + l) * D_MODEL + c];
    fused[b * 2 * D_MODEL + t] = s * (1.f / L_SZ);
}

__global__ void fc1_kernel(const float* __restrict__ fused, const float* __restrict__ w,
                           const float* __restrict__ b, float* __restrict__ o)
{
    int bb = blockIdx.x, t = threadIdx.x;
    float acc = b[t];
    for (int k = 0; k < 2 * D_MODEL; ++k) acc += fused[bb * 2 * D_MODEL + k] * w[k * FC_H + t];
    o[bb * FC_H + t] = fmaxf(acc, 0.f);
}

__global__ void fc2_kernel(const float* __restrict__ a, const float* __restrict__ w,
                           const float* __restrict__ b, float* __restrict__ out)
{
    int bb = blockIdx.x, t = threadIdx.x;
    if (t < NCLS) {
        float acc = b[t];
        for (int k = 0; k < FC_H; ++k) acc += a[bb * FC_H + k] * w[k * NCLS + t];
        out[bb * NCLS + t] = acc;
    }
}

// ---------------------------------------------------------------------------
extern "C" void kernel_launch(void* const* d_in, const int* in_sizes, int n_in,
                              void* d_out, int out_size, void* d_ws, size_t ws_size,
                              hipStream_t stream)
{
    (void)in_sizes; (void)n_in; (void)out_size; (void)ws_size;
    const float* x            = (const float*)d_in[0];
    const float* mamba_proj_w = (const float*)d_in[1];
    const float* mamba_proj_b = (const float*)d_in[2];
    const float* in_proj_w    = (const float*)d_in[3];
    const float* conv_w       = (const float*)d_in[4];
    const float* conv_b       = (const float*)d_in[5];
    const float* x_proj_w     = (const float*)d_in[6];
    const float* dt_proj_w    = (const float*)d_in[7];
    const float* dt_proj_b    = (const float*)d_in[8];
    const float* A_log        = (const float*)d_in[9];
    const float* Dp           = (const float*)d_in[10];
    const float* out_proj_w   = (const float*)d_in[11];
    const float* lstm_w_ih    = (const float*)d_in[12];
    const float* lstm_w_hh    = (const float*)d_in[13];
    const float* lstm_b       = (const float*)d_in[14];
    const float* ca_w         = (const float*)d_in[15];
    const float* ca_b         = (const float*)d_in[16];
    const float* ln_g         = (const float*)d_in[17];
    const float* ln_b         = (const float*)d_in[18];
    const float* fc1_w        = (const float*)d_in[19];
    const float* fc1_b        = (const float*)d_in[20];
    const float* fc2_w        = (const float*)d_in[21];
    const float* fc2_b        = (const float*)d_in[22];
    float* out = (float*)d_out;

    const int M = B_SZ * L_SZ; // 8192 rows
    float* ws = (float*)d_ws;
    size_t o = 0;
    float* u     = ws + o; o += (size_t)M * D_MODEL;
    float* xz    = ws + o; o += (size_t)M * 2 * D_INNER;
    float* xc    = ws + o; o += (size_t)M * D_INNER;
    float* dbl   = ws + o; o += (size_t)M * DBL_W;
    float* delta = ws + o; o += (size_t)M * D_INNER;
    float* yz    = ws + o; o += (size_t)M * D_INNER;
    float* H_M   = ws + o; o += (size_t)M * D_MODEL;
    float* H_L   = ws + o; o += (size_t)M * D_MODEL;
    float* Qb    = ws + o; o += (size_t)M * D_MODEL;
    float* Kb    = ws + o; o += (size_t)M * D_MODEL;
    float* Vb    = ws + o; o += (size_t)M * D_MODEL;
    float* attn  = ws + o; o += (size_t)M * D_MODEL;
    float* tmp   = ws + o; o += (size_t)M * D_MODEL;
    float* H_Lu  = ws + o; o += (size_t)M * D_MODEL;
    float* H_Mu  = ws + o; o += (size_t)M * D_MODEL;
    float* fused = ws + o; o += (size_t)B_SZ * 2 * D_MODEL;
    float* fc1o  = ws + o; o += (size_t)B_SZ * FC_H;

    // f16 staging region (32B aligned: all prior counts are multiples of 16)
    _Float16* h16 = (_Float16*)(ws + o);
    size_t ho = 0;
    _Float16* inW_h  = h16 + ho; ho += (size_t)512 * 128;
    _Float16* xpW_h  = h16 + ho; ho += (size_t)64 * 256;
    _Float16* dtW_h  = h16 + ho; ho += (size_t)256 * 32;
    _Float16* outW_h = h16 + ho; ho += (size_t)128 * 256;
    _Float16* caW_h  = h16 + ho; ho += (size_t)8 * 128 * 128;
    _Float16* Kh     = h16 + ho; ho += (size_t)M * D_MODEL;
    _Float16* Vt     = h16 + ho; ho += (size_t)M * D_MODEL;

    dim3 w32(32);
    const int mt = M / 16; // 512 row tiles
    const size_t WSZ = (size_t)D_MODEL * D_MODEL;

    // --- weight packing (cheap, once per launch) ---
    pack_w_kernel<<<(512 * 128 + 255) / 256, 256, 0, stream>>>(in_proj_w, inW_h, 128, 512, 128, 512);
    pack_w_kernel<<<(64 * 256 + 255) / 256, 256, 0, stream>>>(x_proj_w, xpW_h, 256, 40, 256, 64);
    pack_w_kernel<<<(256 * 32 + 255) / 256, 256, 0, stream>>>(dt_proj_w, dtW_h, 8, 256, 32, 256);
    pack_w_kernel<<<(128 * 256 + 255) / 256, 256, 0, stream>>>(out_proj_w, outW_h, 256, 128, 256, 128);
    for (int j = 0; j < 8; ++j)
        pack_w_kernel<<<(128 * 128 + 255) / 256, 256, 0, stream>>>(ca_w + j * WSZ, caW_h + j * WSZ,
                                                                   128, 128, 128, 128);

    // --- Mamba branch ---
    mproj_kernel<<<M, D_MODEL, 0, stream>>>(x, mamba_proj_w, mamba_proj_b, u);
    wmma_gemm_kernel<<<dim3(8, mt), w32, 0, stream>>>(u, inW_h, nullptr, nullptr, xz, 128, 128, 512, 0);
    conv_kernel<<<(M * D_INNER + 255) / 256, 256, 0, stream>>>(xz, conv_w, conv_b, xc);
    wmma_gemm_kernel<<<dim3(1, mt), w32, 0, stream>>>(xc, xpW_h, nullptr, nullptr, dbl, 256, 256, DBL_W, 0);
    wmma_gemm_kernel<<<dim3(4, mt), w32, 0, stream>>>(dbl, dtW_h, dt_proj_b, nullptr, delta, 32, DBL_W, 256, 2);
    scan_kernel<<<(B_SZ * D_INNER + 255) / 256, 256, 0, stream>>>(delta, xc, dbl, xz, A_log, Dp, yz);
    wmma_gemm_kernel<<<dim3(2, mt), w32, 0, stream>>>(yz, outW_h, nullptr, nullptr, H_M, 256, 256, 128, 0);

    // --- LSTM branch ---
    lstm_kernel<<<B_SZ, 4 * LSTM_H, 0, stream>>>(x, lstm_w_ih, lstm_w_hh, lstm_b, H_L);

    // --- Cross attention dir 0: q = H_L, kv = H_M ---
    wmma_gemm_kernel<<<dim3(2, mt), w32, 0, stream>>>(H_L, caW_h + 0 * WSZ, ca_b + 0 * D_MODEL, nullptr, Qb, 128, 128, 128, 0);
    wmma_gemm_kernel<<<dim3(2, mt), w32, 0, stream>>>(H_M, caW_h + 1 * WSZ, ca_b + 1 * D_MODEL, nullptr, Kb, 128, 128, 128, 0);
    wmma_gemm_kernel<<<dim3(2, mt), w32, 0, stream>>>(H_M, caW_h + 2 * WSZ, ca_b + 2 * D_MODEL, nullptr, Vb, 128, 128, 128, 0);
    f16cvt_kernel<<<(M * D_MODEL + 255) / 256, 256, 0, stream>>>(Kb, Kh, M * D_MODEL);
    vtrans_kernel<<<(M * D_MODEL + 255) / 256, 256, 0, stream>>>(Vb, Vt);
    attn_kernel<<<dim3(L_SZ / 16, NHEAD, B_SZ), w32, 0, stream>>>(Qb, Kh, Vt, attn);
    wmma_gemm_kernel<<<dim3(2, mt), w32, 0, stream>>>(attn, caW_h + 3 * WSZ, ca_b + 3 * D_MODEL, H_L, tmp, 128, 128, 128, 0);
    ln_kernel<<<M, D_MODEL, 0, stream>>>(tmp, ln_g, ln_b, H_Lu);

    // --- Cross attention dir 1: q = H_M, kv = H_L ---
    wmma_gemm_kernel<<<dim3(2, mt), w32, 0, stream>>>(H_M, caW_h + 4 * WSZ, ca_b + 4 * D_MODEL, nullptr, Qb, 128, 128, 128, 0);
    wmma_gemm_kernel<<<dim3(2, mt), w32, 0, stream>>>(H_L, caW_h + 5 * WSZ, ca_b + 5 * D_MODEL, nullptr, Kb, 128, 128, 128, 0);
    wmma_gemm_kernel<<<dim3(2, mt), w32, 0, stream>>>(H_L, caW_h + 6 * WSZ, ca_b + 6 * D_MODEL, nullptr, Vb, 128, 128, 128, 0);
    f16cvt_kernel<<<(M * D_MODEL + 255) / 256, 256, 0, stream>>>(Kb, Kh, M * D_MODEL);
    vtrans_kernel<<<(M * D_MODEL + 255) / 256, 256, 0, stream>>>(Vb, Vt);
    attn_kernel<<<dim3(L_SZ / 16, NHEAD, B_SZ), w32, 0, stream>>>(Qb, Kh, Vt, attn);
    wmma_gemm_kernel<<<dim3(2, mt), w32, 0, stream>>>(attn, caW_h + 7 * WSZ, ca_b + 7 * D_MODEL, H_M, tmp, 128, 128, 128, 0);
    ln_kernel<<<M, D_MODEL, 0, stream>>>(tmp, ln_g + D_MODEL, ln_b + D_MODEL, H_Mu);

    // --- Head ---
    pool_kernel<<<B_SZ, 2 * D_MODEL, 0, stream>>>(H_Mu, H_Lu, fused);
    fc1_kernel<<<B_SZ, FC_H, 0, stream>>>(fused, fc1_w, fc1_b, fc1o);
    fc2_kernel<<<B_SZ, 32, 0, stream>>>(fc1o, fc2_w, fc2_b, out);
}
